// AttentionModel_59614146068701
// MI455X (gfx1250) — matrix operand order repaired and checked by reading, not compiled, not run on hardware
//
#include <hip/hip_runtime.h>
#include <hip/hip_bf16.h>

typedef __attribute__((ext_vector_type(16))) _Float16 v16h;
typedef __attribute__((ext_vector_type(8)))  _Float16 v8h;
typedef __attribute__((ext_vector_type(4)))  _Float16 v4h;
typedef __attribute__((ext_vector_type(8)))  float    v8f;
typedef unsigned int u32x4 __attribute__((ext_vector_type(4)));
typedef int          i32x8 __attribute__((ext_vector_type(8)));
typedef int          i32x4 __attribute__((ext_vector_type(4)));

#define F_DIM   512
#define SEQ_L   1024
#define N_HEADS 8
#define HEAD_D  64
#define ROWS    8192        // N * L
#define VOCAB   1024
#define NL      4
#define NEGINF  (-1.0e30f)

#define BSTAGE_K     128                 // K-halves per TDM stage
#define BROW_HALFS   136                 // 128 data + 8 pad halves (16B TDM pad)

// Assemble a 16-half WMMA fragment from two contiguous 16-byte chunks.
__device__ __forceinline__ v16h load2x8(const _Float16* p0, const _Float16* p1) {
  v8h lo = *(const v8h*)p0;
  v8h hi = *(const v8h*)p1;
  v16h r;
#pragma unroll
  for (int e = 0; e < 8; ++e) { r[e] = lo[e]; r[e + 8] = hi[e]; }
  return r;
}

// ---------------------------------------------------------------------------
// TDM: async-copy a 2D tile (rows x rowElems f16, global row stride strideElems)
// into LDS with 16B padding appended per 256B of data (bank-conflict-free rows).
__device__ __forceinline__ void tdm_load_tile(const _Float16* gsrc, void* ldst,
                                              int rowElems, int rows,
                                              int strideElems) {
  unsigned long long ga = (unsigned long long)(uintptr_t)gsrc;
  unsigned int la = (unsigned int)(uintptr_t)ldst;   // low 32 bits = LDS offset
  u32x4 g0;
  g0[0] = 1u;                                        // count = 1 (valid)
  g0[1] = la;                                        // lds_addr
  g0[2] = (unsigned int)ga;                          // global_addr[31:0]
  g0[3] = ((unsigned int)(ga >> 32) & 0x01ffffffu) | (2u << 30);  // [56:32] | type=2
  i32x8 g1;
  // data_size=1 (2B) | pad_enable | pad_interval=5 (64 dwords) | pad_amount=3 (4 dwords)
  g1[0] = (1 << 16) | (1 << 20) | (5 << 22) | (3 << 25);
  g1[1] = (rowElems & 0xffff) << 16;                 // tensor_dim0[15:0]
  g1[2] = ((rowElems >> 16) & 0xffff) | ((rows & 0xffff) << 16);  // td0 hi | td1 lo
  g1[3] = ((rows >> 16) & 0xffff) | ((rowElems & 0xffff) << 16);  // td1 hi | tile_dim0
  g1[4] = rows & 0xffff;                             // tile_dim1 (tile_dim2 = 0)
  g1[5] = strideElems;                               // tensor_dim0_stride[31:0]
  g1[6] = 0;
  g1[7] = 0;
  i32x4 z4 = {0, 0, 0, 0};
  i32x8 z8 = {0, 0, 0, 0, 0, 0, 0, 0};
  __builtin_amdgcn_tensor_load_to_lds(g0, g1, z4, z4, z8, 0);
}

// ---------------------------------------------------------------------------
// f32 (K x N, per layer) -> f16 transposed (N x K, per layer)
__global__ void __launch_bounds__(256)
cvtT_kernel(const float* __restrict__ src, _Float16* __restrict__ dst,
            int K, int N, int total) {
  int idx = blockIdx.x * 256 + threadIdx.x;
  if (idx >= total) return;
  int l = idx / (K * N);
  int r = idx % (K * N);
  int n = r / K, k = r % K;
  dst[idx] = (_Float16)src[(size_t)l * K * N + (size_t)k * N + n];
}

// ---------------------------------------------------------------------------
// embedding gather -> x (f32 residual) and xh (f16 activations), 4 elems/thread
__global__ void __launch_bounds__(256)
embed_kernel(const int* __restrict__ tokens, const float* __restrict__ emb,
             float* __restrict__ x, _Float16* __restrict__ xh) {
  size_t idx = ((size_t)blockIdx.x * 256 + threadIdx.x) * 4;
  if (idx >= (size_t)ROWS * F_DIM) return;
  int row = (int)(idx >> 9);
  int f   = (int)(idx & 511);
  float4 v = *(const float4*)(emb + (size_t)tokens[row] * F_DIM + f);
  *(float4*)(x + idx) = v;
  v4h h = { (_Float16)v.x, (_Float16)v.y, (_Float16)v.z, (_Float16)v.w };
  *(v4h*)(xh + idx) = h;
}

// ---------------------------------------------------------------------------
// WMMA GEMM: C(MxN) = A(MxK,f16 row-major) * Bt(NxK,f16 transposed) [+bias][ReLU]
// All 8 waves of a block share one 64-column B strip; B tiles are TDM-copied
// into LDS (double-buffered, padded rows), A is read via global_load_b128.
// MODE 0: f32 row-major   MODE 1: f16 row-major
// MODE 2: f16 per-head (b,h,L,64)   MODE 3: f16 per-head transposed (b,h,64,L)
template <int MODE, bool RELU>
__global__ void __launch_bounds__(256)
gemm_wmma(const _Float16* __restrict__ A, const _Float16* __restrict__ Bt,
          const float* __restrict__ bias, float* __restrict__ outF,
          _Float16* __restrict__ outH, int M, int N, int K) {
  __shared__ _Float16 bstage[2][64 * BROW_HALFS];
  const int wid   = threadIdx.x >> 5;
  const int lane  = threadIdx.x & 31;
  const int nlane = lane & 15;
  const int kh    = lane >> 4;
  const int tiles_m = M >> 4;
  const int tm    = (blockIdx.x * 8 + wid) % tiles_m;
  const int tn    = (blockIdx.x * 8) / tiles_m;       // uniform across the block
  const int row0  = tm * 16;
  const int col0  = tn * 64;
  const int nstage = K / BSTAGE_K;

  if (wid == 0)
    tdm_load_tile(Bt + (size_t)col0 * K, &bstage[0][0], BSTAGE_K, 64, K);

  v8f acc[4] = {};
  for (int s = 0; s < nstage; ++s) {
    if (wid == 0) {
      if (s + 1 < nstage) {
        tdm_load_tile(Bt + (size_t)col0 * K + (s + 1) * BSTAGE_K,
                      &bstage[(s + 1) & 1][0], BSTAGE_K, 64, K);
        __builtin_amdgcn_s_wait_tensorcnt(1);   // stage s done; s+1 may be in flight
      } else {
        __builtin_amdgcn_s_wait_tensorcnt(0);   // last stage: drain everything
      }
    }
    __syncthreads();                       // stage s data visible to all waves
    const _Float16* bb = &bstage[s & 1][0];
#pragma unroll
    for (int ktl = 0; ktl < BSTAGE_K; ktl += 32) {
      const int kt = s * BSTAGE_K + ktl;
      const _Float16* Ap = A + (size_t)(row0 + nlane) * K + kt + 8 * kh;
      v16h a = load2x8(Ap, Ap + 16);
#pragma unroll
      for (int t = 0; t < 4; ++t) {
        const _Float16* Bp = bb + (t * 16 + nlane) * BROW_HALFS + ktl + 8 * kh;
        v16h b = load2x8(Bp, Bp + 16);
        acc[t] = __builtin_amdgcn_wmma_f32_16x16x32_f16(false, a, false, b,
                                                        (short)0, acc[t], false, false);
      }
    }
    __syncthreads();                       // buffer (s&1) free for TDM overwrite
  }
#pragma unroll
  for (int t = 0; t < 4; ++t) {
#pragma unroll
    for (int v = 0; v < 8; ++v) {
      int m   = v + 8 * kh;               // C layout: M = vgpr + 8*(lane>=16)
      int row = row0 + m;
      int col = col0 + t * 16 + nlane;
      float val = acc[t][v];
      if (bias) val += bias[col];
      if (RELU) val = fmaxf(val, 0.f);
      if (MODE == 0) {
        outF[(size_t)row * N + col] = val;
      } else if (MODE == 1) {
        outH[(size_t)row * N + col] = (_Float16)val;
      } else if (MODE == 2) {           // (b,h,L,64)
        int bb2 = row >> 10, i = row & 1023, h = col >> 6, d = col & 63;
        outH[((((size_t)bb2 * N_HEADS + h) << 10) + i) * HEAD_D + d] = (_Float16)val;
      } else {                          // MODE 3: (b,h,64,L)
        int bb2 = row >> 10, i = row & 1023, h = col >> 6, d = col & 63;
        outH[(((size_t)bb2 * N_HEADS + h) * HEAD_D + d) * SEQ_L + i] = (_Float16)val;
      }
    }
  }
}

// ---------------------------------------------------------------------------
// Flash-style attention.  score[i,j] = sum_d K[i,d]*V[j,d], causal mask j<=i,
// softmax over j, O[i,:] = sum_j P[i,j]*Q[j,:].  One wave per (b,h, 16-row tile).
// K,V in (b,h,L,64); Q in transposed (b,h,64,L) so every fragment is B128 loads.
__global__ void __launch_bounds__(256)
attn_kernel(const _Float16* __restrict__ qT, const _Float16* __restrict__ khp,
            const _Float16* __restrict__ vh, _Float16* __restrict__ attn) {
  __shared__ _Float16 lds[8][16 * 32];   // per-wave P staging tile (C->A relayout)
  const int wid   = threadIdx.x >> 5;
  const int lane  = threadIdx.x & 31;
  const int gwave = blockIdx.x * 8 + wid;
  const int it    = gwave & 63;          // L/16 = 64 row tiles
  const int bh    = gwave >> 6;          // b*8 + h
  const int i0    = it * 16;
  const int nlane = lane & 15;
  const int khf   = lane >> 4;
  const _Float16* Qt = qT  + (size_t)bh * SEQ_L * HEAD_D;  // (64, L)
  const _Float16* K  = khp + (size_t)bh * SEQ_L * HEAD_D;  // (L, 64)
  const _Float16* V  = vh  + (size_t)bh * SEQ_L * HEAD_D;  // (L, 64)

  const _Float16* kr = K + (size_t)(i0 + nlane) * HEAD_D + 8 * khf;
  v16h aK0 = load2x8(kr, kr + 16);
  v16h aK1 = load2x8(kr + 32, kr + 48);

  v8f O[4] = {};
  float mrow[8], lrow[8];
#pragma unroll
  for (int v = 0; v < 8; ++v) { mrow[v] = NEGINF; lrow[v] = 0.f; }

  for (int jb = 0; jb <= i0 + 15; jb += 32) {
    float S[2][8];
#pragma unroll
    for (int t = 0; t < 2; ++t) {
      int jt = jb + 16 * t;
      if (jt > i0 + 15) {
#pragma unroll
        for (int v = 0; v < 8; ++v) S[t][v] = NEGINF;
      } else {
        const _Float16* vr = V + (size_t)(jt + nlane) * HEAD_D + 8 * khf;
        v16h b0 = load2x8(vr, vr + 16);
        v16h b1 = load2x8(vr + 32, vr + 48);
        v8f c = {};
        c = __builtin_amdgcn_wmma_f32_16x16x32_f16(false, aK0, false, b0, (short)0, c, false, false);
        c = __builtin_amdgcn_wmma_f32_16x16x32_f16(false, aK1, false, b1, (short)0, c, false, false);
        bool full = (jt + 15) <= i0;     // whole tile below diagonal
#pragma unroll
        for (int v = 0; v < 8; ++v) {
          float s = c[v];
          if (!full && (jt + nlane > i0 + v + 8 * khf)) s = NEGINF;  // j > i
          S[t][v] = s;
        }
      }
    }
    float mnew[8], alpha[8];
#pragma unroll
    for (int v = 0; v < 8; ++v) {
      float tm = fmaxf(S[0][v], S[1][v]);
#pragma unroll
      for (int off = 1; off < 16; off <<= 1) tm = fmaxf(tm, __shfl_xor(tm, off, 16));
      mnew[v] = fmaxf(mrow[v], tm);
    }
#pragma unroll
    for (int t = 0; t < 2; ++t)
#pragma unroll
      for (int v = 0; v < 8; ++v) {
        float p = __expf(S[t][v] - mnew[v]);
        S[t][v] = p;
        lds[wid][(v + 8 * khf) * 32 + t * 16 + nlane] = (_Float16)p;
      }
#pragma unroll
    for (int v = 0; v < 8; ++v) {
      float s = S[0][v] + S[1][v];
#pragma unroll
      for (int off = 1; off < 16; off <<= 1) s += __shfl_xor(s, off, 16);
      alpha[v] = __expf(mrow[v] - mnew[v]);
      lrow[v]  = lrow[v] * alpha[v] + s;
      mrow[v]  = mnew[v];
    }
#pragma unroll
    for (int db = 0; db < 4; ++db)
#pragma unroll
      for (int v = 0; v < 8; ++v) O[db][v] *= alpha[v];

    const _Float16* lp = &lds[wid][nlane * 32 + 8 * khf];
    v16h aP = load2x8(lp, lp + 16);
#pragma unroll
    for (int db = 0; db < 4; ++db) {
      const _Float16* qp = Qt + (size_t)(db * 16 + nlane) * SEQ_L + jb + 8 * khf;
      v16h bq = load2x8(qp, qp + 16);
      O[db] = __builtin_amdgcn_wmma_f32_16x16x32_f16(false, aP, false, bq,
                                                     (short)0, O[db], false, false);
    }
  }

  const int b = bh >> 3, h = bh & 7;
#pragma unroll
  for (int db = 0; db < 4; ++db)
#pragma unroll
    for (int v = 0; v < 8; ++v) {
      int m = v + 8 * khf;
      float val = O[db][v] / lrow[v];
      attn[((size_t)(b * SEQ_L + i0 + m)) * F_DIM + h * HEAD_D + db * 16 + nlane] =
          (_Float16)val;
    }
}

// ---------------------------------------------------------------------------
// fused residual + LayerNorm: x = LN(x + h2)*gamma + beta ; also emit f16 copy
__global__ void __launch_bounds__(256)
ln_kernel(const float* __restrict__ h2, const float* __restrict__ gamma,
          const float* __restrict__ beta, float* __restrict__ x,
          _Float16* __restrict__ xh) {
  const int wid  = threadIdx.x >> 5;
  const int lane = threadIdx.x & 31;
  const int row  = blockIdx.x * 8 + wid;
  const float* hp = h2 + (size_t)row * F_DIM;
  float*       xp = x  + (size_t)row * F_DIM;
  float y[16];
  float sum = 0.f;
#pragma unroll
  for (int e = 0; e < 16; ++e) {
    float t = xp[lane + 32 * e] + hp[lane + 32 * e];
    y[e] = t;
    sum += t;
  }
#pragma unroll
  for (int off = 1; off < 32; off <<= 1) sum += __shfl_xor(sum, off, 32);
  float mu = sum * (1.f / F_DIM);
  float var = 0.f;
#pragma unroll
  for (int e = 0; e < 16; ++e) { float d = y[e] - mu; var += d * d; }
#pragma unroll
  for (int off = 1; off < 32; off <<= 1) var += __shfl_xor(var, off, 32);
  float r = rsqrtf(var * (1.f / F_DIM) + 1e-5f);
#pragma unroll
  for (int e = 0; e < 16; ++e) {
    int f = lane + 32 * e;
    float o = (y[e] - mu) * r * gamma[f] + beta[f];
    xp[f] = o;
    xh[(size_t)row * F_DIM + f] = (_Float16)o;
  }
}

// ---------------------------------------------------------------------------
extern "C" void kernel_launch(void* const* d_in, const int* in_sizes, int n_in,
                              void* d_out, int out_size, void* d_ws, size_t ws_size,
                              hipStream_t stream) {
  (void)in_sizes; (void)n_in; (void)out_size; (void)ws_size;
  const int*   tokens = (const int*)d_in[0];
  const float* embed  = (const float*)d_in[1];
  const float* Wq     = (const float*)d_in[2];
  const float* Wk     = (const float*)d_in[3];
  const float* Wv     = (const float*)d_in[4];
  const float* W1     = (const float*)d_in[5];
  const float* b1     = (const float*)d_in[6];
  const float* W2     = (const float*)d_in[7];
  const float* b2     = (const float*)d_in[8];
  const float* gamma  = (const float*)d_in[9];
  const float* beta   = (const float*)d_in[10];
  const float* Wout   = (const float*)d_in[11];
  const float* bout   = (const float*)d_in[12];

  char* p = (char*)d_ws;
  auto alloc = [&](size_t bytes) { char* r = p; p += (bytes + 255) & ~(size_t)255; return r; };
  const size_t ACT = (size_t)ROWS * F_DIM;
  float*    x    = (float*)   alloc(ACT * 4);
  _Float16* xh   = (_Float16*)alloc(ACT * 2);
  _Float16* qT   = (_Float16*)alloc(ACT * 2);   // (b,h,64,L)
  _Float16* kh   = (_Float16*)alloc(ACT * 2);   // (b,h,L,64)
  _Float16* vh   = (_Float16*)alloc(ACT * 2);   // (b,h,L,64)
  _Float16* attn = (_Float16*)alloc(ACT * 2);
  _Float16* h1   = (_Float16*)alloc(ACT * 2);
  float*    h2   = (float*)   alloc(ACT * 4);
  const size_t WSZ = (size_t)NL * F_DIM * F_DIM;   // per weight family
  _Float16* wqT   = (_Float16*)alloc(WSZ * 2);
  _Float16* wkT   = (_Float16*)alloc(WSZ * 2);
  _Float16* wvT   = (_Float16*)alloc(WSZ * 2);
  _Float16* w1T   = (_Float16*)alloc(WSZ * 2);
  _Float16* w2T   = (_Float16*)alloc(WSZ * 2);
  _Float16* woutT = (_Float16*)alloc((size_t)F_DIM * VOCAB * 2);

  // weight conversion + transpose (Bt layout: N x K per layer)
  const int nW = NL * F_DIM * F_DIM;          // 1,048,576
  cvtT_kernel<<<nW / 256, 256, 0, stream>>>(Wq, wqT, F_DIM, F_DIM, nW);
  cvtT_kernel<<<nW / 256, 256, 0, stream>>>(Wk, wkT, F_DIM, F_DIM, nW);
  cvtT_kernel<<<nW / 256, 256, 0, stream>>>(Wv, wvT, F_DIM, F_DIM, nW);
  cvtT_kernel<<<nW / 256, 256, 0, stream>>>(W1, w1T, F_DIM, F_DIM, nW);
  cvtT_kernel<<<nW / 256, 256, 0, stream>>>(W2, w2T, F_DIM, F_DIM, nW);
  const int nWo = F_DIM * VOCAB;              // 524,288
  cvtT_kernel<<<nWo / 256, 256, 0, stream>>>(Wout, woutT, F_DIM, VOCAB, nWo);

  // embedding
  embed_kernel<<<(int)(ACT / 1024), 256, 0, stream>>>(tokens, embed, x, xh);

  const int gGemm512 = (ROWS / 16) * (F_DIM / 64) / 8;   // 512 blocks
  const int gGemmOut = (ROWS / 16) * (VOCAB / 64) / 8;   // 1024 blocks
  const int gAttn    = (8 * N_HEADS * (SEQ_L / 16)) / 8; // 512 blocks

  for (int l = 0; l < NL; ++l) {
    const size_t wo = (size_t)l * F_DIM * F_DIM;
    gemm_wmma<3, false><<<gGemm512, 256, 0, stream>>>(xh, wqT + wo, nullptr, nullptr, qT,
                                                      ROWS, F_DIM, F_DIM);
    gemm_wmma<2, false><<<gGemm512, 256, 0, stream>>>(xh, wkT + wo, nullptr, nullptr, kh,
                                                      ROWS, F_DIM, F_DIM);
    gemm_wmma<2, false><<<gGemm512, 256, 0, stream>>>(xh, wvT + wo, nullptr, nullptr, vh,
                                                      ROWS, F_DIM, F_DIM);
    attn_kernel<<<gAttn, 256, 0, stream>>>(qT, kh, vh, attn);
    gemm_wmma<1, true><<<gGemm512, 256, 0, stream>>>(attn, w1T + wo, b1 + l * F_DIM,
                                                     nullptr, h1, ROWS, F_DIM, F_DIM);
    gemm_wmma<0, false><<<gGemm512, 256, 0, stream>>>(h1, w2T + wo, b2 + l * F_DIM,
                                                      h2, nullptr, ROWS, F_DIM, F_DIM);
    ln_kernel<<<ROWS / 8, 256, 0, stream>>>(h2, gamma + l * F_DIM, beta + l * F_DIM, x, xh);
  }

  gemm_wmma<0, false><<<gGemmOut, 256, 0, stream>>>(xh, woutT, bout, (float*)d_out,
                                                    nullptr, ROWS, VOCAB, F_DIM);
}
